// roiVGGYoloWithGradCam_77103252898340
// MI455X (gfx1250) — matrix operand order repaired and checked by reading, not compiled, not run on hardware
//
#include <hip/hip_runtime.h>
#include <hip/hip_bf16.h>

// ---------------------------------------------------------------------------
// MI455X (gfx1250). FC1 (256x25088x4096) dominates: W1 = 411 MB fp32 streamed
// from HBM every call -> 17.6 us floor at 23.3 TB/s; arithmetic intensity is
// 2*M = 512 FLOP per 4B weight, so bf16 WMMA (16x16x32, f32 accum) is needed
// to stay at the bandwidth floor. Bytes-in-flight comes from a double-buffered
// LDS pipeline + global_load_async_to_lds_b128 for the bf16 activations +
// global_prefetch_b8 at distance-3 tiles for the weight stream.
// ---------------------------------------------------------------------------

typedef __attribute__((ext_vector_type(16))) __bf16          v16bf;
typedef __attribute__((ext_vector_type(8)))  float           v8f;
typedef __attribute__((ext_vector_type(16))) unsigned short  v16u;
typedef __attribute__((ext_vector_type(4)))  unsigned short  u16x4;

union FragU { v16u v; u16x4 q[4]; };

__device__ __forceinline__ unsigned rne_bf_hi(float x) {
  unsigned u = __float_as_uint(x);
  return u + 0x7FFFu + ((u >> 16) & 1u);   // bf16(x) in bits [31:16]
}

__device__ __forceinline__ unsigned short f2bf_rne(float x) {
  return (unsigned short)(rne_bf_hi(x) >> 16);
}

// pack {bf16(lo), bf16(hi)} into one dword: v_perm_b32 merge of the two
// rounded high halves; bytes [a2,a3,b2,b3] with src1=a (bytes 0-3), src0=b.
__device__ __forceinline__ unsigned pack_bf16x2(float lo, float hi) {
  unsigned a = rne_bf_hi(lo);
  unsigned b = rne_bf_hi(hi);
  return __builtin_amdgcn_perm(b, a, 0x07060302u);
}

// CDNA5 async copy: LDS[vdst + b] = MEM[vaddr + b], tracked by ASYNCcnt.
// Generic LDS pointers hold the LDS byte offset in their low 32 bits.
__device__ __forceinline__ void async_copy_b128(unsigned lds_off,
                                                const void* gptr) {
  asm volatile("global_load_async_to_lds_b128 %0, %1, off"
               :: "v"(lds_off), "v"((unsigned long long)gptr)
               : "memory");
}

__device__ __forceinline__ void wait_async0() {
#if __has_builtin(__builtin_amdgcn_s_wait_asynccnt)
  __builtin_amdgcn_s_wait_asynccnt(0);
#else
  asm volatile("s_wait_asynccnt 0x0" ::: "memory");
#endif
}

// ---------------------------------------------------------------------------
// ROI max pool (torchvision roi_pool, spatial_scale=1, 7x7) -> bf16 rows
// ---------------------------------------------------------------------------
__global__ void roi_pool_bf16_kernel(const float* __restrict__ feat,
                                     const int*   __restrict__ boxes,
                                     const int*   __restrict__ box_img,
                                     unsigned short* __restrict__ xout,
                                     int NB, int C, int H, int W) {
  int idx = blockIdx.x * blockDim.x + threadIdx.x;
  int total = NB * C * 49;
  if (idx >= total) return;

  int pw = idx % 7;
  int ph = (idx / 7) % 7;
  int c  = (idx / 49) % C;
  int n  = idx / (49 * C);

  int x1 = boxes[n * 4 + 0], y1 = boxes[n * 4 + 1];
  int x2 = boxes[n * 4 + 2], y2 = boxes[n * 4 + 3];
  float bin_w = fmaxf((float)(x2 - x1 + 1), 1.0f) * (1.0f / 7.0f);
  float bin_h = fmaxf((float)(y2 - y1 + 1), 1.0f) * (1.0f / 7.0f);

  int ws = min(max((int)floorf((float)pw * bin_w) + x1, 0), W);
  int we = min(max((int)ceilf((float)(pw + 1) * bin_w) + x1, 0), W);
  int hs = min(max((int)floorf((float)ph * bin_h) + y1, 0), H);
  int he = min(max((int)ceilf((float)(ph + 1) * bin_h) + y1, 0), H);

  const float* fb = feat + ((size_t)box_img[n] * C + c) * (size_t)(H * W);
  float m = -__builtin_inff();
  for (int h = hs; h < he; ++h)
    for (int w = ws; w < we; ++w)
      m = fmaxf(m, fb[h * W + w]);
  if (!(we > ws && he > hs)) m = 0.0f;          // where(isfinite, ., 0)

  xout[(size_t)n * (size_t)(C * 49) + (size_t)c * 49 + ph * 7 + pw] = f2bf_rne(m);
}

// ---------------------------------------------------------------------------
// C = act(A[M,K](bf16) @ W[K,N](f32 stream -> bf16) + bias)
// Block tile 128x64, BK=32, double-buffered LDS pipeline:
//   iter i: async A copy + W loads for tile i+1, WMMA tile i,
//           packed cvt/store W tile i+1, s_wait_asynccnt, barrier.
// 8 waves: 2(M:64 rows) x 4(N:16 cols), 4 x v_wmma_f32_16x16x32_bf16 / K-step.
// ---------------------------------------------------------------------------
template <bool RELU, bool OUT_BF16>
__global__ __launch_bounds__(256)
void gemm_bf16_wmma_kernel(const unsigned short* __restrict__ A,
                           const float* __restrict__ Wm,
                           const float* __restrict__ bias,
                           void* __restrict__ outp,
                           int M, int N, int K) {
  constexpr int BM = 128, BN = 64, BK = 32, BKP = BK + 8; // 80B rows, 16B-aligned
  __shared__ unsigned short As[2][BM][BKP];
  __shared__ unsigned short Bs[2][BN][BKP];

  const int tid   = threadIdx.x;
  const int lane  = tid & 31;
  const int wave  = tid >> 5;     // 0..7 (wave32)
  const int wm    = wave >> 2;    // 0..1 -> 64 rows
  const int wn    = wave & 3;     // 0..3 -> 16 cols
  const int lm    = lane & 15;
  const int lhalf = lane >> 4;

  const int m0 = blockIdx.y * BM;
  const int n0 = blockIdx.x * BN;

  // A-tile copy coords: 32B (16 halves) per thread
  const int arow = tid >> 1;            // 0..127
  const int acol = (tid & 1) << 4;      // 0 or 16 halves
  // W-tile coords: two adjacent K rows, 4 columns per thread
  const int kp = tid >> 4;              // 0..15 -> k = 2*kp
  const int n4 = (tid & 15) << 2;       // 0..60 step 4

  // fragment coordinates (direct LDS indexing keeps these as ds_load_b128)
  const int aRow = wm * 64 + lm;        // + t*16
  const int aCol = lhalf * 8;
  const int bRow = wn * 16 + lm;
  const int bCol = lhalf * 8;

  v8f acc[4] = {};

  // running global pointers
  const char*  agp = (const char*)(A + (size_t)(m0 + arow) * K + acol);
  const float* wgp = Wm + (size_t)(2 * kp) * N + n0 + n4;
  const size_t wstep = (size_t)BK * N;                  // floats per K-tile

  // async-copy LDS destinations (plain 32-bit offsets, safe to index)
  const unsigned ldsA[2] = {
    (unsigned)(unsigned long long)(const void*)&As[0][arow][acol],
    (unsigned)(unsigned long long)(const void*)&As[1][arow][acol] };

  const int NT = K / BK;

  // ---- prologue: stage tile 0 into buffer 0 --------------------------------
  async_copy_b128(ldsA[0],      agp);
  async_copy_b128(ldsA[0] + 16, agp + 16);
  {
    float4 wa = *(const float4*)wgp;
    float4 wb = *(const float4*)(wgp + N);
    *(unsigned*)&Bs[0][n4 + 0][2 * kp] = pack_bf16x2(wa.x, wb.x);
    *(unsigned*)&Bs[0][n4 + 1][2 * kp] = pack_bf16x2(wa.y, wb.y);
    *(unsigned*)&Bs[0][n4 + 2][2 * kp] = pack_bf16x2(wa.z, wb.z);
    *(unsigned*)&Bs[0][n4 + 3][2 * kp] = pack_bf16x2(wa.w, wb.w);
  }
  agp += 2 * BK;          // BK halves in bytes
  wgp += wstep;
  wait_async0();
  __syncthreads();

  for (int it = 0; it < NT; ++it) {
    const int  cur  = it & 1;
    const int  nxt  = cur ^ 1;
    const bool more = (it + 1 < NT);

    float4 wa, wb;
    if (more) {
      async_copy_b128(ldsA[nxt],      agp);   // ASYNCcnt path, bypasses VGPRs
      async_copy_b128(ldsA[nxt] + 16, agp + 16);
      wa = *(const float4*)wgp;               // loadcnt wait lands after WMMAs
      wb = *(const float4*)(wgp + N);
      if (it + 3 < NT) {                      // GL2 prefetch, distance 3 tiles
        __builtin_prefetch(wgp + 2 * wstep,     0, 1);
        __builtin_prefetch(wgp + 2 * wstep + N, 0, 1);
      }
      agp += 2 * BK;
      wgp += wstep;
    }

    // ---- compute from buffer `cur` ----------------------------------------
    // B frag (lane<16 -> K{0-7,16-23}, lane>=16 -> K{8-15,24-31})
    FragU bfr;
    {
      const unsigned short* bp = &Bs[cur][bRow][bCol];
      bfr.q[0] = *(const u16x4*)(bp);
      bfr.q[1] = *(const u16x4*)(bp + 4);
      bfr.q[2] = *(const u16x4*)(bp + 16);
      bfr.q[3] = *(const u16x4*)(bp + 20);
    }
    // preload all 4 A frags (lane<16 -> K{0-7,16-23}, lane>=16 -> K{8-15,24-31})
    FragU af[4];
#pragma unroll
    for (int t = 0; t < 4; ++t) {
      const unsigned short* ap = &As[cur][aRow + t * 16][aCol];
      af[t].q[0] = *(const u16x4*)(ap);
      af[t].q[1] = *(const u16x4*)(ap + 4);
      af[t].q[2] = *(const u16x4*)(ap + 16);
      af[t].q[3] = *(const u16x4*)(ap + 20);
    }
#pragma unroll
    for (int t = 0; t < 4; ++t) {
      acc[t] = __builtin_amdgcn_wmma_f32_16x16x32_bf16(
          false, __builtin_bit_cast(v16bf, af[t].v),
          false, __builtin_bit_cast(v16bf, bfr.v),
          (short)0, acc[t], false, false);
    }

    if (more) {                       // packed bf16 pairs -> ds_store_b32
      *(unsigned*)&Bs[nxt][n4 + 0][2 * kp] = pack_bf16x2(wa.x, wb.x);
      *(unsigned*)&Bs[nxt][n4 + 1][2 * kp] = pack_bf16x2(wa.y, wb.y);
      *(unsigned*)&Bs[nxt][n4 + 2][2 * kp] = pack_bf16x2(wa.z, wb.z);
      *(unsigned*)&Bs[nxt][n4 + 3][2 * kp] = pack_bf16x2(wa.w, wb.w);
    }
    wait_async0();                    // As[nxt] resident before next iter
    __syncthreads();
  }

  // ---- epilogue: C/D layout lane<16 -> M=v, lane>=16 -> M=v+8 --------------
  const int   gcol = n0 + wn * 16 + lm;
  const float bv   = bias[gcol];
#pragma unroll
  for (int t = 0; t < 4; ++t) {
#pragma unroll
    for (int v = 0; v < 8; ++v) {
      int   grow = m0 + wm * 64 + t * 16 + lhalf * 8 + v;
      float val  = acc[t][v] + bv;
      if (RELU) val = fmaxf(val, 0.0f);
      if (OUT_BF16)
        ((unsigned short*)outp)[(size_t)grow * N + gcol] = f2bf_rne(val);
      else
        ((float*)outp)[(size_t)grow * N + gcol] = val;
    }
  }
}

// ---------------------------------------------------------------------------
// fmri[b, j] = sum_{n : box_img[n]==b} y[n, j]
// ---------------------------------------------------------------------------
__global__ void segsum_kernel(const float* __restrict__ y,
                              const int*   __restrict__ box_img,
                              float* __restrict__ out,
                              int NB, int NSEG, int D) {
  int idx = blockIdx.x * blockDim.x + threadIdx.x;
  if (idx >= NSEG * D) return;
  int b = idx / D;
  int j = idx - b * D;
  float s = 0.0f;
  for (int n = 0; n < NB; ++n)
    if (box_img[n] == b) s += y[(size_t)n * D + j];
  out[idx] = s;
}

// ---------------------------------------------------------------------------
extern "C" void kernel_launch(void* const* d_in, const int* in_sizes, int n_in,
                              void* d_out, int out_size, void* d_ws, size_t ws_size,
                              hipStream_t stream) {
  const float* feat    = (const float*)d_in[0];   // [64,512,14,14]
  const int*   boxes   = (const int*)  d_in[1];   // [256,4]
  const int*   box_img = (const int*)  d_in[2];   // [256]
  const float* W1      = (const float*)d_in[3];   // [25088,4096]
  const float* b1      = (const float*)d_in[4];
  const float* W2      = (const float*)d_in[5];   // [4096,1024]
  const float* b2      = (const float*)d_in[6];
  const float* W3      = (const float*)d_in[7];   // [1024,1024]
  const float* b3      = (const float*)d_in[8];

  const int B = 64, C = 512, H = 14, W = 14;
  const int NB = 256;                 // boxes
  const int K1 = 25088, N1 = 4096, N2 = 1024, N3 = 1024;

  // workspace layout (bf16 activations + fp32 y), all 16B aligned
  unsigned short* x_bf = (unsigned short*)d_ws;              // 256 x 25088
  unsigned short* h1   = x_bf + (size_t)NB * K1;             // 256 x 4096
  unsigned short* h2   = h1   + (size_t)NB * N1;             // 256 x 1024
  float*          y    = (float*)(h2 + (size_t)NB * N2);     // 256 x 1024

  // 1) ROI max pool -> bf16 rows
  {
    int total  = NB * C * 49;
    int blocks = (total + 255) / 256;
    roi_pool_bf16_kernel<<<blocks, 256, 0, stream>>>(feat, boxes, box_img,
                                                     x_bf, NB, C, H, W);
  }
  // 2) FC1 + ReLU (weight-streaming bound: 411 MB fp32 -> bf16 WMMA)
  gemm_bf16_wmma_kernel<true, true>
      <<<dim3(N1 / 64, NB / 128), 256, 0, stream>>>(x_bf, W1, b1, h1, NB, N1, K1);
  // 3) FC2 + ReLU
  gemm_bf16_wmma_kernel<true, true>
      <<<dim3(N2 / 64, NB / 128), 256, 0, stream>>>(h1, W2, b2, h2, NB, N2, N1);
  // 4) FC3 (fp32 out)
  gemm_bf16_wmma_kernel<false, false>
      <<<dim3(N3 / 64, NB / 128), 256, 0, stream>>>(h2, W3, b3, (void*)y, NB, N3, N2);
  // 5) segment sum -> d_out [64,1024]
  {
    int total  = B * N3;
    int blocks = (total + 255) / 256;
    segsum_kernel<<<blocks, 256, 0, stream>>>(y, box_img, (float*)d_out, NB, B, N3);
  }
}
